// RGCNLinkPredict_8693013807309
// MI455X (gfx1250) — compile-verified, matrix operand order checked by reading, not run on hardware
//
#include <hip/hip_runtime.h>
#include <hip/hip_bf16.h>
#include <math.h>

// ---------------------------------------------------------------------------
// RGCN 2-layer forward for MI455X (gfx1250), wave32 + v_wmma_f32_16x16x32_bf16
// GEMM keeps the A row-strip in registers and sweeps all 17 weight matrices.
// ---------------------------------------------------------------------------

#define N_NODES 50000
#define N_EDGES 800000
#define N_RELS  16
#define DIM     128
#define MTILES  (N_NODES / 16)   // 3125, exact

typedef __attribute__((ext_vector_type(16))) __bf16 v16bf;
typedef __attribute__((ext_vector_type(8)))  float  v8f;

static __device__ __forceinline__ unsigned short f32_to_bf16(float f) {
    unsigned int u = __float_as_uint(f);
    // round-to-nearest-even
    unsigned int r = (u + 0x7FFFu + ((u >> 16) & 1u)) >> 16;
    return (unsigned short)r;
}

// ---------------------------------------------------------------------------
// Pack h rows into the WMMA 16x32 bf16 A-matrix per-lane layout.
// Ab layout: [mt][kk][lane][16 halves], flat index = ((mt*4+kk)*32+lane)*16
// lane<16:  row = mt*16+lane,    K = kk*32 + {0..7, 16..23}
// lane>=16: row = mt*16+lane-16, K = kk*32 + {8..15, 24..31}
// ---------------------------------------------------------------------------
__global__ void pack_A_kernel(const float* __restrict__ h,
                              unsigned short* __restrict__ Ab) {
    int t = blockIdx.x * 256 + threadIdx.x;       // one thread per (mt,kk,lane)
    if (t >= MTILES * 128) return;
    int mt   = t >> 7;
    int rem  = t & 127;
    int kk   = rem >> 5;
    int lane = rem & 31;
    int row   = mt * 16 + (lane & 15);
    int kbase = kk * 32 + ((lane >> 4) << 3);
    const float* hp = h + (size_t)row * DIM + kbase;
    unsigned short* op = Ab + (size_t)t * 16;
#pragma unroll
    for (int j = 0; j < 8; ++j) {
        op[j]     = f32_to_bf16(hp[j]);
        op[j + 8] = f32_to_bf16(hp[j + 16]);
    }
}

// ---------------------------------------------------------------------------
// Pack the 16 relation weights + self-loop weight (as "relation 16") into the
// WMMA 32x16 bf16 B-matrix per-lane layout (B[k][n], lane = column n group).
// Wb flat index = (((r*8+nt)*4+kk)*32+lane)*16
// ---------------------------------------------------------------------------
__global__ void pack_W_kernel(const float* __restrict__ W,      // [16,128,128]
                              const float* __restrict__ loop_w, // [128,128]
                              unsigned short* __restrict__ Wb) {
    int t = blockIdx.x * 256 + threadIdx.x;       // 17*8*4*32 = 17408 threads
    if (t >= 17 * 1024) return;
    int r    = t >> 10;
    int rem  = t & 1023;
    int nt   = rem >> 7;
    int rem2 = rem & 127;
    int kk   = rem2 >> 5;
    int lane = rem2 & 31;
    int col   = nt * 16 + (lane & 15);
    int kbase = kk * 32 + ((lane >> 4) << 3);
    const float* wsrc = (r < N_RELS) ? (W + (size_t)r * DIM * DIM) : loop_w;
    unsigned short* op = Wb + (size_t)t * 16;
#pragma unroll
    for (int j = 0; j < 8; ++j) {
        op[j]     = f32_to_bf16(wsrc[(size_t)(kbase + j) * DIM + col]);
        op[j + 8] = f32_to_bf16(wsrc[(size_t)(kbase + j + 16) * DIM + col]);
    }
}

// ---------------------------------------------------------------------------
// Gate: gate[r][n] = sigmoid(h[n] . gate_w[r]) ; one wave per node.
// ---------------------------------------------------------------------------
__global__ void gate_kernel(const float* __restrict__ h,
                            const float* __restrict__ gw,   // [16,128]
                            float* __restrict__ gate) {     // [16,N]
    int node = (blockIdx.x * 256 + threadIdx.x) >> 5;
    int lane = threadIdx.x & 31;
    if (node >= N_NODES) return;
    const float* hp = h + (size_t)node * DIM;
    float x0 = hp[lane], x1 = hp[lane + 32], x2 = hp[lane + 64], x3 = hp[lane + 96];
#pragma unroll
    for (int r = 0; r < N_RELS; ++r) {
        const float* g = gw + r * DIM;
        float p = x0 * g[lane] + x1 * g[lane + 32] + x2 * g[lane + 64] + x3 * g[lane + 96];
#pragma unroll
        for (int off = 16; off > 0; off >>= 1) p += __shfl_down(p, off);
        if (lane == 0)
            gate[(size_t)r * N_NODES + node] = 1.0f / (1.0f + __expf(-p));
    }
}

// ---------------------------------------------------------------------------
// WMMA GEMM: grid = 3125 row-tiles; 8 waves/block, wave w owns column tile
// [16w, 16w+16).  A row-strip is loaded ONCE into registers (32 VGPRs) and
// reused across all 16 relations + self-loop => 68 WMMAs per wave, A operand
// traffic cut 17x.  r<16 writes h_all[r]; self-loop initializes accum with
// h@loop_w + bias.
// ---------------------------------------------------------------------------
__global__ void rgcn_wmma_gemm_kernel(const unsigned short* __restrict__ Ab,
                                      const unsigned short* __restrict__ Wb,
                                      const float* __restrict__ bias,
                                      float* __restrict__ h_all,   // [16,N,128]
                                      float* __restrict__ accum) { // [N,128]
    int mt   = blockIdx.x;
    int w    = threadIdx.x >> 5;      // column tile 0..7
    int lane = threadIdx.x & 31;

    const unsigned short* Ap = Ab + (size_t)mt * 128 * 16;
    v16bf a0 = *(const v16bf*)(Ap + (size_t)(0 * 32 + lane) * 16);
    v16bf a1 = *(const v16bf*)(Ap + (size_t)(1 * 32 + lane) * 16);
    v16bf a2 = *(const v16bf*)(Ap + (size_t)(2 * 32 + lane) * 16);
    v16bf a3 = *(const v16bf*)(Ap + (size_t)(3 * 32 + lane) * 16);

    int col  = w * 16 + (lane & 15);
    int row0 = mt * 16 + ((lane >> 4) << 3);   // C layout: vgpr v -> M = row0+v

    // 16 relation GEMMs, A register-resident
    for (int r = 0; r < N_RELS; ++r) {
        const unsigned short* Bp = Wb + (size_t)(r * 8 + w) * 128 * 16;
        v16bf b0 = *(const v16bf*)(Bp + (size_t)(0 * 32 + lane) * 16);
        v16bf b1 = *(const v16bf*)(Bp + (size_t)(1 * 32 + lane) * 16);
        v16bf b2 = *(const v16bf*)(Bp + (size_t)(2 * 32 + lane) * 16);
        v16bf b3 = *(const v16bf*)(Bp + (size_t)(3 * 32 + lane) * 16);
        v8f c = {};
        c = __builtin_amdgcn_wmma_f32_16x16x32_bf16(false, a0, false, b0, (short)0, c, false, false);
        c = __builtin_amdgcn_wmma_f32_16x16x32_bf16(false, a1, false, b1, (short)0, c, false, false);
        c = __builtin_amdgcn_wmma_f32_16x16x32_bf16(false, a2, false, b2, (short)0, c, false, false);
        c = __builtin_amdgcn_wmma_f32_16x16x32_bf16(false, a3, false, b3, (short)0, c, false, false);
        float* o = h_all + ((size_t)r * N_NODES + row0) * DIM + col;
#pragma unroll
        for (int v = 0; v < 8; ++v) o[(size_t)v * DIM] = c[v];
    }

    // Self-loop GEMM (weight index 16): accum = h @ loop_w + bias
    {
        const unsigned short* Bp = Wb + (size_t)(N_RELS * 8 + w) * 128 * 16;
        v16bf b0 = *(const v16bf*)(Bp + (size_t)(0 * 32 + lane) * 16);
        v16bf b1 = *(const v16bf*)(Bp + (size_t)(1 * 32 + lane) * 16);
        v16bf b2 = *(const v16bf*)(Bp + (size_t)(2 * 32 + lane) * 16);
        v16bf b3 = *(const v16bf*)(Bp + (size_t)(3 * 32 + lane) * 16);
        v8f c = {};
        c = __builtin_amdgcn_wmma_f32_16x16x32_bf16(false, a0, false, b0, (short)0, c, false, false);
        c = __builtin_amdgcn_wmma_f32_16x16x32_bf16(false, a1, false, b1, (short)0, c, false, false);
        c = __builtin_amdgcn_wmma_f32_16x16x32_bf16(false, a2, false, b2, (short)0, c, false, false);
        c = __builtin_amdgcn_wmma_f32_16x16x32_bf16(false, a3, false, b3, (short)0, c, false, false);
        float badd = bias[col];
        float* o = accum + (size_t)row0 * DIM + col;
#pragma unroll
        for (int v = 0; v < 8; ++v) o[(size_t)v * DIM] = c[v] + badd;
    }
}

// ---------------------------------------------------------------------------
// Edge kernel: one wave per edge. msg = h_all[rel,src] * (norm * gate), then
// atomic scatter-add into accum[dst] (accum is L2-resident, 25.6 MB).
// ---------------------------------------------------------------------------
__global__ void edge_kernel(const float* __restrict__ h_all,
                            const float* __restrict__ gate,
                            const float* __restrict__ norm,
                            const int* __restrict__ src,
                            const int* __restrict__ dst,
                            const int* __restrict__ rel,
                            float* __restrict__ accum) {
    int e    = blockIdx.x * 8 + (threadIdx.x >> 5);
    int lane = threadIdx.x & 31;
    if (e >= N_EDGES) return;
    int s  = src[e];
    int d  = dst[e];
    int rr = rel[e];
    float coeff = norm[e] * gate[(size_t)rr * N_NODES + s];
    const float4* mp = (const float4*)(h_all + ((size_t)rr * N_NODES + s) * DIM);
    float4 m = mp[lane];                       // 512 contiguous bytes per edge
    float* ap = accum + (size_t)d * DIM + lane * 4;
    atomicAdd(ap + 0, m.x * coeff);
    atomicAdd(ap + 1, m.y * coeff);
    atomicAdd(ap + 2, m.z * coeff);
    atomicAdd(ap + 3, m.w * coeff);
}

__global__ void relu_inplace_kernel(float* __restrict__ x, int n) {
    int t = blockIdx.x * 256 + threadIdx.x;
    if (t < n) x[t] = fmaxf(x[t], 0.0f);
}

// ---------------------------------------------------------------------------
// Launch
// ---------------------------------------------------------------------------
extern "C" void kernel_launch(void* const* d_in, const int* in_sizes, int n_in,
                              void* d_out, int out_size, void* d_ws, size_t ws_size,
                              hipStream_t stream) {
    const float* h       = (const float*)d_in[0];
    const float* norm    = (const float*)d_in[1];
    const float* weight0 = (const float*)d_in[2];
    const float* bias0   = (const float*)d_in[3];
    const float* loop_w0 = (const float*)d_in[4];
    const float* gate_w0 = (const float*)d_in[5];
    const float* weight1 = (const float*)d_in[6];
    const float* bias1   = (const float*)d_in[7];
    const float* loop_w1 = (const float*)d_in[8];
    const float* gate_w1 = (const float*)d_in[9];
    const int*   src     = (const int*)d_in[10];
    const int*   dst     = (const int*)d_in[11];
    const int*   rel     = (const int*)d_in[12];
    float*       out     = (float*)d_out;

    // Workspace layout (all 256B aligned):
    constexpr size_t AB_BYTES   = (size_t)MTILES * 4 * 32 * 16 * 2;       // 12.8 MB
    constexpr size_t WB_BYTES   = (size_t)17 * 8 * 4 * 32 * 16 * 2;       // 0.56 MB
    constexpr size_t GATE_BYTES = (size_t)N_RELS * N_NODES * 4;           // 3.2 MB
    constexpr size_t ACC_BYTES  = (size_t)N_NODES * DIM * 4;              // 25.6 MB
    char* w = (char*)d_ws;
    unsigned short* Ab   = (unsigned short*)(w);
    unsigned short* Wb   = (unsigned short*)(w + AB_BYTES);
    float*          gate = (float*)(w + AB_BYTES + WB_BYTES);
    float*          acc  = (float*)(w + AB_BYTES + WB_BYTES + GATE_BYTES);
    float*          hall = (float*)(w + AB_BYTES + WB_BYTES + GATE_BYTES + ACC_BYTES);

    dim3 blk(256);
    dim3 gemm_grid(MTILES);
    int packA_blocks = (MTILES * 128 + 255) / 256;
    int packW_blocks = (17 * 1024 + 255) / 256;
    int gate_blocks  = (N_NODES + 7) / 8;       // one wave per node, 8 waves/blk
    int edge_blocks  = N_EDGES / 8;             // one wave per edge, 8 waves/blk
    int nd_elems     = N_NODES * DIM;
    int relu_blocks  = (nd_elems + 255) / 256;

    // ---------------- Layer 0 (input h, output -> acc, ReLU) ----------------
    pack_W_kernel<<<packW_blocks, blk, 0, stream>>>(weight0, loop_w0, Wb);
    pack_A_kernel<<<packA_blocks, blk, 0, stream>>>(h, Ab);
    gate_kernel<<<gate_blocks, blk, 0, stream>>>(h, gate_w0, gate);
    rgcn_wmma_gemm_kernel<<<gemm_grid, blk, 0, stream>>>(Ab, Wb, bias0, hall, acc);
    edge_kernel<<<edge_blocks, blk, 0, stream>>>(hall, gate, norm, src, dst, rel, acc);
    relu_inplace_kernel<<<relu_blocks, blk, 0, stream>>>(acc, nd_elems);

    // ---------------- Layer 1 (input acc, output -> d_out, no ReLU) ---------
    pack_W_kernel<<<packW_blocks, blk, 0, stream>>>(weight1, loop_w1, Wb);
    pack_A_kernel<<<packA_blocks, blk, 0, stream>>>(acc, Ab);
    gate_kernel<<<gate_blocks, blk, 0, stream>>>(acc, gate_w1, gate);
    rgcn_wmma_gemm_kernel<<<gemm_grid, blk, 0, stream>>>(Ab, Wb, bias1, hall, out);
    edge_kernel<<<edge_blocks, blk, 0, stream>>>(hall, gate, norm, src, dst, rel, out);
}
